// CrossAttention_18339510354565
// MI455X (gfx1250) — matrix operand order repaired and checked by reading, not compiled, not run on hardware
//
#include <hip/hip_runtime.h>
#include <hip/hip_bf16.h>
#include <math.h>
#include <stdint.h>

// MI455X / gfx1250, wave32.
// Matrix math: V_WMMA_F32_16X16X32_BF16 (bf16 in, fp32 accumulate).
// Data movement: double-buffered Tensor Data Mover DMA (tensor_load_to_lds,
// pipelined with s_wait_tensorcnt <=1) + register-prefetch for fp32 staging.

typedef __bf16 bf16;
typedef __attribute__((ext_vector_type(16))) __bf16 v16bf;
typedef __attribute__((ext_vector_type(8)))  __bf16 v8bf;
typedef __attribute__((ext_vector_type(8)))  float  v8f;
typedef __attribute__((ext_vector_type(4)))  unsigned int u32x4;
typedef __attribute__((ext_vector_type(8)))  int i32x8;
typedef __attribute__((ext_vector_type(4)))  int i32x4;

#define EMBED  1024
#define HIDDEN 512
#define SEQ    4096
#define BATCH  4
#define NROWS  (BATCH*SEQ)   // 16384

__device__ __forceinline__ v16bf cat8(v8bf a, v8bf b){
  return __builtin_shufflevector(a, b, 0,1,2,3,4,5,6,7,8,9,10,11,12,13,14,15);
}
__device__ __forceinline__ v8f wmma_bf16(v16bf a, v16bf b, v8f c){
  return __builtin_amdgcn_wmma_f32_16x16x32_bf16(false, a, false, b, (short)0, c, false, false);
}
// immediate-arg wrapper: wait for all TDM ops (0) or all but newest (1)
__device__ __forceinline__ void wait_tensor(bool keep_one_in_flight){
  if (keep_one_in_flight) __builtin_amdgcn_s_wait_tensorcnt(1);
  else                    __builtin_amdgcn_s_wait_tensorcnt(0);
}

// ---------------- Tensor Data Mover: 2D tile global->LDS ----------------
// D# group0/group1 per ISA 8.3/8.4: count=1, data_size=2B, type=2, LDS pad.
__device__ __forceinline__ void tdm_load_2d(
    unsigned lds_byte_addr, const void* gaddr,
    unsigned tile_d0, unsigned tile_d1,            // tile size (elems, rows)
    unsigned tensor_d0, unsigned tensor_d1,        // tensor size (elems, rows)
    unsigned long long d0_stride_elems,            // row stride in elems
    unsigned pad_interval_code, unsigned pad_amount_code)
{
  unsigned long long ga = (unsigned long long)(uintptr_t)gaddr;
  u32x4 g0;
  g0[0] = 1u;                                          // count=1 (valid), user mode
  g0[1] = lds_byte_addr;                               // LDS dest (bytes)
  g0[2] = (unsigned)(ga & 0xffffffffu);                // global_addr[31:0]
  g0[3] = (unsigned)((ga >> 32) & 0x1ffffffu) | (2u << 30);  // [56:32] | type=2
  i32x8 g1;
  unsigned w0 = (1u << 16)                             // data_size = 2 bytes
              | (1u << 20)                             // pad_enable
              | (pad_interval_code << 22)
              | (pad_amount_code << 25);
  g1[0] = (int)w0;
  g1[1] = (int)((tensor_d0 & 0xffffu) << 16);          // abar_addr=0 | dim0[15:0]
  g1[2] = (int)((tensor_d0 >> 16) | ((tensor_d1 & 0xffffu) << 16));
  g1[3] = (int)((tensor_d1 >> 16) | (tile_d0 << 16));  // dim1[31:16] | tile_dim0
  g1[4] = (int)(tile_d1 & 0xffffu);                    // tile_dim1 | tile_dim2=0
  g1[5] = (int)(unsigned)(d0_stride_elems & 0xffffffffu);
  g1[6] = (int)(unsigned)((d0_stride_elems >> 32) & 0xffffu);
  g1[7] = 0;
  i32x4 z4 = {0,0,0,0};
  i32x8 z8 = {0,0,0,0,0,0,0,0};
  __builtin_amdgcn_tensor_load_to_lds(g0, g1, z4, z4, z8, 0);
}
__device__ __forceinline__ unsigned lds_off(const void* p){
  return (unsigned)(uintptr_t)p;   // generic VA low 32 bits = LDS byte offset
}

// ---------------- fragment loads (ISA 7.12.2 layouts) ----------------
__device__ __forceinline__ v16bf load_a_frag(const bf16* base, int lda, int m0, int k0){
  int lane = threadIdx.x & 31;
  const bf16* p = base + (size_t)(m0 + (lane & 15)) * lda + k0 + 8*(lane >> 4);
  v8bf x0 = *(const v8bf*)p;
  v8bf x1 = *(const v8bf*)(p + 16);
  return cat8(x0, x1);
}
__device__ __forceinline__ v16bf load_b_frag(const bf16* base, int ldb, int k0, int n0){
  int lane = threadIdx.x & 31;
  const bf16* p = base + (size_t)(n0 + (lane & 15)) * ldb + k0 + 8*(lane >> 4);
  v8bf x0 = *(const v8bf*)p;
  v8bf x1 = *(const v8bf*)(p + 16);
  return cat8(x0, x1);
}

// ---------------- generic GEMM: C[M,N] = A[M,K] * W[K,N] (+bias)(+resid) ----------------
// WG tile 64x128, 8 waves of 32x32, K-step 32.
// fp32 A: register-prefetch staging. bf16 A: double-buffered TDM DMA.
template<typename AT, bool BIAS, bool RES, bool OUT_BF16, bool STORE_T>
__global__ __launch_bounds__(256) void gemm_kernel(
    const AT* __restrict__ A, const float* __restrict__ W,
    const float* __restrict__ bias, const float* __restrict__ resid,
    float* __restrict__ Cf, bf16* __restrict__ Cb, int M, int N, int K)
{
  __shared__ bf16 As[2*64*40];  // 2 buffers [64 rows][32 k + pad] (fp32 path uses buf0)
  __shared__ bf16 Bt[128*40];   // n-major [128 n][32 k + pad]
  const int m0 = blockIdx.y * 64, n0 = blockIdx.x * 128;
  const int w = threadIdx.x >> 5, lane = threadIdx.x & 31;
  const int wm = w >> 2, wn = w & 3;
  const int hi = lane >> 4, cn = lane & 15;
  const int arow = threadIdx.x >> 2, ac8 = threadIdx.x & 3;
  constexpr bool A_F32 = (sizeof(AT) == 4);

  v8f acc[2][2] = {};
  float4 ra0, ra1;        // fp32-A prefetch regs
  float4 rb[4];           // W prefetch regs

  if constexpr (A_F32){
    const float* g = (const float*)A + (size_t)(m0 + arow) * K + ac8*8;
    ra0 = *(const float4*)g; ra1 = *(const float4*)(g + 4);
  } else {
    if (w == 0)   // prologue: DMA A chunk 0 into buffer 0
      tdm_load_2d(lds_off(As), (const bf16*)A + (size_t)m0 * K,
                  32, 64, (unsigned)K, (unsigned)M, (unsigned)K, 3, 3);
  }
  for (int i = 0; i < 4; ++i){
    int c4 = threadIdx.x*4 + i, kk = c4 >> 5, n4 = c4 & 31;
    rb[i] = *(const float4*)(W + (size_t)kk * N + n0 + n4*4);
  }

  for (int k0 = 0; k0 < K; k0 += 32){
    const int it = k0 >> 5;
    const int kn = k0 + 32;
    __syncthreads();                 // previous compute done reading LDS
    if constexpr (A_F32){
      v8bf o;
      o[0]=(bf16)ra0.x; o[1]=(bf16)ra0.y; o[2]=(bf16)ra0.z; o[3]=(bf16)ra0.w;
      o[4]=(bf16)ra1.x; o[5]=(bf16)ra1.y; o[6]=(bf16)ra1.z; o[7]=(bf16)ra1.w;
      *(v8bf*)(As + arow*40 + ac8*8) = o;
    } else {
      if (w == 0 && kn < K)          // DMA next A chunk into other buffer
        tdm_load_2d(lds_off(As + ((it+1)&1)*64*40),
                    (const bf16*)A + (size_t)m0 * K + kn,
                    32, 64, (unsigned)K, (unsigned)M, (unsigned)K, 3, 3);
    }
    for (int i = 0; i < 4; ++i){     // W fp32 -> bf16, transposed scatter
      int c4 = threadIdx.x*4 + i, kk = c4 >> 5, n4 = c4 & 31;
      Bt[(n4*4+0)*40 + kk] = (bf16)rb[i].x;
      Bt[(n4*4+1)*40 + kk] = (bf16)rb[i].y;
      Bt[(n4*4+2)*40 + kk] = (bf16)rb[i].z;
      Bt[(n4*4+3)*40 + kk] = (bf16)rb[i].w;
    }
    if (kn < K){                     // prefetch next W tile (overlaps compute)
      if constexpr (A_F32){
        const float* g = (const float*)A + (size_t)(m0 + arow) * K + kn + ac8*8;
        ra0 = *(const float4*)g; ra1 = *(const float4*)(g + 4);
      }
      for (int i = 0; i < 4; ++i){
        int c4 = threadIdx.x*4 + i, kk = c4 >> 5, n4 = c4 & 31;
        rb[i] = *(const float4*)(W + (size_t)(kn + kk) * N + n0 + n4*4);
      }
    }
    if constexpr (!A_F32){           // oldest DMA (this chunk) done; next stays in flight
      if (w == 0) wait_tensor(kn < K);
    }
    __syncthreads();
    const bf16* Ab = A_F32 ? As : (As + (it&1)*64*40);
    v16bf a0 = load_a_frag(Ab, 40, wm*32,      0);
    v16bf a1 = load_a_frag(Ab, 40, wm*32 + 16, 0);
    v16bf b0 = load_b_frag(Bt, 40, 0, wn*32);
    v16bf b1 = load_b_frag(Bt, 40, 0, wn*32 + 16);
    acc[0][0] = wmma_bf16(a0, b0, acc[0][0]);
    acc[0][1] = wmma_bf16(a0, b1, acc[0][1]);
    acc[1][0] = wmma_bf16(a1, b0, acc[1][0]);
    acc[1][1] = wmma_bf16(a1, b1, acc[1][1]);
  }

  for (int ti = 0; ti < 2; ++ti)
    for (int tj = 0; tj < 2; ++tj){
      int gn = n0 + wn*32 + tj*16 + cn;
      float bia = BIAS ? bias[gn] : 0.0f;
      for (int r = 0; r < 8; ++r){
        int gm = m0 + wm*32 + ti*16 + r + 8*hi;
        float v = acc[ti][tj][r] + bia;
        if (RES) v += resid[(size_t)gm * N + gn];
        if (OUT_BF16){
          if (STORE_T){ int bb = gm >> 12, s = gm & (SEQ-1);
            Cb[((size_t)bb * N + gn) * SEQ + s] = (bf16)v;
          } else Cb[(size_t)gm * N + gn] = (bf16)v;
        } else Cf[(size_t)gm * N + gn] = v;
      }
    }
}

// ---------------- flash attention ----------------
// 32 query rows per WG, 64-wide KV tiles, online softmax.
// All tiles via double-buffered TDM DMA (pipelined s_wait_tensorcnt<=1).
// LDS aliasing (disjoint lifetimes): K double-buffer <-> Ps;  Sld <-> V double-buffer.
__global__ __launch_bounds__(256) void flash_kernel(
    const bf16* __restrict__ Q, const bf16* __restrict__ Kmat,
    const bf16* __restrict__ Vt, bf16* __restrict__ Ctx)
{
  __shared__ bf16  Qs[32*520];                     // 33,280 B  [q][512 h + pad]
  __shared__ bf16  KsPs[2*64*72];                  // 18,432 B  K dbl-buf | Ps alias
  __shared__ __align__(16) char smemSV[2*64*40*2]; // 10,240 B  Sld | V dbl-buf
  __shared__ float rowM[32], rowL[32], rowScale[32];
  __shared__ float red[32*8], red2[32*8];
  bf16*  Ks0 = KsPs;
  bf16*  Ks1 = KsPs + 64*72;
  bf16*  Ps  = KsPs;                   // alias K buffer 0 (S phase vs PV phase)
  float* Sld = (float*)smemSV;         // alias V buffers (dead before first V DMA)
  bf16*  Vt0 = (bf16*)smemSV;
  bf16*  Vt1 = (bf16*)smemSV + 64*40;

  const int tid = threadIdx.x;
  const int w = tid >> 5, lane = tid & 31;
  const int wm = w >> 2, wn = w & 3;
  const int hi = lane >> 4, cn = lane & 15;
  const int qb = blockIdx.x, b = blockIdx.y;

  const bf16* Qg = Q    + ((size_t)b * SEQ + qb*32) * HIDDEN;
  const bf16* Kg = Kmat + (size_t)b * SEQ * HIDDEN;
  const bf16* Vg = Vt   + (size_t)b * HIDDEN * SEQ;     // [h][s]

  // Q block via TDM: 512-elem rows (256 DW -> code 7) + 16B pad -> stride 520
  if (w == 0){
    tdm_load_2d(lds_off(Qs), Qg, 512, 32, 512, SEQ, HIDDEN, 7, 3);
    __builtin_amdgcn_s_wait_tensorcnt(0);
  }
  if (tid < 32){ rowM[tid] = -1e30f; rowL[tid] = 0.0f; }
  __syncthreads();

  v8f ctx[8] = {};   // 16 rows x 128 cols/wave: h = t*64 + wn*16 + cn
  const float SCALE = 0.044194173824159216f;   // 1/sqrt(512)

  for (int j = 0; j < SEQ/64; ++j){
    v8f sacc = {};
    const bf16* Kj = Kg + (size_t)j*64*HIDDEN;
    // ---- S = Q*K^T: 8 pipelined 64-wide h chunks (rows 32 DW -> code 4) ----
    __syncthreads();                       // prev j readers of Ps/V buffers done
    if (w == 0)
      tdm_load_2d(lds_off(Ks0), Kj, 64, 64, HIDDEN, SEQ, HIDDEN, 4, 3);
    for (int c = 0; c < 8; ++c){
      if (w == 0){
        if (c < 7)
          tdm_load_2d(lds_off((c+1)&1 ? Ks1 : Ks0), Kj + (c+1)*64,
                      64, 64, HIDDEN, SEQ, HIDDEN, 4, 3);
        wait_tensor(c < 7);
      }
      __syncthreads();                     // publish chunk c
      const bf16* kb = (c&1) ? Ks1 : Ks0;
      int hc = c*64;
      for (int ks = 0; ks < 64; ks += 32){
        v16bf a  = load_a_frag(Qs, 520, wm*16, hc + ks);
        v16bf bb = load_b_frag(kb, 72, ks, wn*16);   // B(k=h, n=kv) = K[kv][h]
        sacc = wmma_bf16(a, bb, sacc);
      }
      __syncthreads();                     // readers done -> buffer reusable
    }
    for (int r = 0; r < 8; ++r)
      Sld[(wm*16 + r + 8*hi)*66 + wn*16 + cn] = sacc[r] * SCALE;
    __syncthreads();
    // ---- online softmax: 8 threads/row ----
    { int row = tid >> 3, q8 = tid & 7;
      float mloc = -1e30f;
      for (int c2 = 0; c2 < 8; ++c2) mloc = fmaxf(mloc, Sld[row*66 + q8*8 + c2]);
      red[row*8 + q8] = mloc;
      __syncthreads();
      float mt = rowM[row];
      for (int i = 0; i < 8; ++i) mt = fmaxf(mt, red[row*8 + i]);
      float sloc = 0.0f;
      for (int c2 = 0; c2 < 8; ++c2){
        float p = __expf(Sld[row*66 + q8*8 + c2] - mt);
        Ps[row*72 + q8*8 + c2] = (bf16)p;
        sloc += p;
      }
      red2[row*8 + q8] = sloc;
      __syncthreads();
      if (q8 == 0){
        float mold = rowM[row];
        float st = 0.0f;
        for (int i = 0; i < 8; ++i) st += red2[row*8 + i];
        float sc = __expf(mold - mt);
        rowL[row] = rowL[row]*sc + st;
        rowM[row] = mt;
        rowScale[row] = sc;
      }
      __syncthreads();                     // Sld now dead -> V buffers may be written
    }
    // ---- ctx += P*V: 16 pipelined 64-row h chunks (rows 16 DW -> code 3) ----
    if (w == 0)
      tdm_load_2d(lds_off(Vt0), Vg + (size_t)0*SEQ + j*64,
                  32, 64, SEQ, HIDDEN, SEQ, 3, 3);
    for (int r = 0; r < 8; ++r){           // rescale ctx (overlaps first V DMA)
      float sc = rowScale[wm*16 + r + 8*hi];
      for (int t = 0; t < 8; ++t) ctx[t][r] *= sc;
    }
    for (int u = 0; u < 16; ++u){
      if (w == 0){
        if (u < 15){
          int un = u + 1, nh = un >> 3, ncn = un & 7;
          tdm_load_2d(lds_off((un&1) ? Vt1 : Vt0),
                      Vg + (size_t)(ncn*64)*SEQ + j*64 + nh*32,
                      32, 64, SEQ, HIDDEN, SEQ, 3, 3);
        }
        wait_tensor(u < 15);
      }
      __syncthreads();                     // publish chunk u
      int half = u >> 3, ch = u & 7;       // h tile = ch*64 + wn*16
      v16bf a  = load_a_frag(Ps, 72, wm*16, half*32);
      v16bf bb = load_b_frag((u&1) ? Vt1 : Vt0, 40, 0, wn*16);
      ctx[ch] = wmma_bf16(a, bb, ctx[ch]);
      __syncthreads();                     // readers done -> buffer reusable
    }
  }
  __syncthreads();
  bf16* Cg = Ctx + ((size_t)b*SEQ + qb*32) * HIDDEN;
  for (int r = 0; r < 8; ++r){
    int rl = wm*16 + r + 8*hi;
    float inv = 1.0f / rowL[rl];
    for (int t = 0; t < 8; ++t){
      int col = t*64 + wn*16 + cn;
      Cg[(size_t)rl*HIDDEN + col] = (bf16)(ctx[t][r] * inv);
    }
  }
}

// ---------------- LayerNorm: one WG per 1024-wide row, in place ----------------
__global__ __launch_bounds__(256) void ln_kernel(float* __restrict__ out,
    const float* __restrict__ gamma, const float* __restrict__ beta)
{
  __shared__ float sb[256];
  const int tid = threadIdx.x;
  float* rowp = out + (size_t)blockIdx.x * EMBED;
  float v[4], s = 0.0f;
  for (int i = 0; i < 4; ++i){ v[i] = rowp[tid + i*256]; s += v[i]; }
  sb[tid] = s; __syncthreads();
  for (int st = 128; st > 0; st >>= 1){
    if (tid < st) sb[tid] += sb[tid + st];
    __syncthreads();
  }
  float mu = sb[0] * (1.0f/1024.0f);
  __syncthreads();
  float s2 = 0.0f;
  for (int i = 0; i < 4; ++i){ float d = v[i] - mu; s2 += d*d; }
  sb[tid] = s2; __syncthreads();
  for (int st = 128; st > 0; st >>= 1){
    if (tid < st) sb[tid] += sb[tid + st];
    __syncthreads();
  }
  float rstd = rsqrtf(sb[0]*(1.0f/1024.0f) + 1e-5f);
  for (int i = 0; i < 4; ++i){
    int col = tid + i*256;
    rowp[col] = (v[i] - mu)*rstd*gamma[col] + beta[col];
  }
}

extern "C" void kernel_launch(void* const* d_in, const int* in_sizes, int n_in,
                              void* d_out, int out_size, void* d_ws, size_t ws_size,
                              hipStream_t stream)
{
  (void)in_sizes; (void)n_in; (void)out_size; (void)ws_size;
  const float* q_feat  = (const float*)d_in[0];
  const float* kv_feat = (const float*)d_in[1];
  const float* Wq  = (const float*)d_in[2];
  const float* Wk  = (const float*)d_in[3];
  const float* Wv  = (const float*)d_in[4];
  const float* Wo  = (const float*)d_in[5];
  const float* bo  = (const float*)d_in[6];
  const float* Wfc = (const float*)d_in[7];
  const float* ln_g = (const float*)d_in[8];
  const float* ln_b = (const float*)d_in[9];
  float* out = (float*)d_out;

  // workspace: Q 16MB | K 16MB | Vt 16MB | ctx 16MB | attn 32MB = 96MB
  char* ws = (char*)d_ws;
  bf16* Qb   = (bf16*)(ws);
  bf16* Kb   = (bf16*)(ws + (size_t)16*1024*1024);
  bf16* Vtb  = (bf16*)(ws + (size_t)32*1024*1024);
  bf16* Ctxb = (bf16*)(ws + (size_t)48*1024*1024);
  bf16* Attn = (bf16*)(ws + (size_t)64*1024*1024);

  dim3 blk(256);
  dim3 gproj(HIDDEN/128, NROWS/64);   // (4, 256)
  gemm_kernel<float,false,false,true,false><<<gproj, blk, 0, stream>>>(q_feat,  Wq, nullptr, nullptr, nullptr, Qb,  NROWS, HIDDEN, EMBED);
  gemm_kernel<float,false,false,true,false><<<gproj, blk, 0, stream>>>(kv_feat, Wk, nullptr, nullptr, nullptr, Kb,  NROWS, HIDDEN, EMBED);
  gemm_kernel<float,false,false,true,true ><<<gproj, blk, 0, stream>>>(kv_feat, Wv, nullptr, nullptr, nullptr, Vtb, NROWS, HIDDEN, EMBED);

  dim3 gflash(SEQ/32, BATCH);
  flash_kernel<<<gflash, blk, 0, stream>>>(Qb, Kb, Vtb, Ctxb);

  dim3 gout(EMBED/128, NROWS/64);     // (8, 256)
  gemm_kernel<bf16,true ,true ,true ,false><<<gout, blk, 0, stream>>>(Ctxb, Wo,  bo,      q_feat,  nullptr, Attn, NROWS, EMBED, HIDDEN);
  gemm_kernel<bf16,false,false,false,false><<<gout, blk, 0, stream>>>(Attn, Wfc, nullptr, nullptr, out,     nullptr, NROWS, EMBED, EMBED);
  ln_kernel<<<dim3(NROWS), blk, 0, stream>>>(out, ln_g, ln_b);
}
